// Wisard_61100204752930
// MI455X (gfx1250) — 8-bit matrix operand order repaired and checked by reading, compiled, not run on hardware
//
#include <hip/hip_runtime.h>


typedef int v8i_t __attribute__((ext_vector_type(8)));

#define ENTRY      1024
#define TUPLE      16
#define NCLASSES   10
#define NNEURONS   64
#define ADDR_BITS  16
#define TS         16
#define BLOCK      128

__global__ __launch_bounds__(BLOCK)
void wisard_wmma_kernel(const int* __restrict__ samples,
                        const int* __restrict__ tmap,
                        const float* __restrict__ ram,
                        float* __restrict__ out)
{
    __shared__ __align__(16) unsigned char sBits[TS * ENTRY];
    __shared__ __align__(16) unsigned char mb[TS * ENTRY];
    __shared__ unsigned short mapl[ENTRY];
    __shared__ float accS[TS];

    const int tid  = threadIdx.x;
    const int lane = tid & 31;
    const int wid  = tid >> 5;
    const int half = lane >> 4;
    const int col  = lane & 15;

    const int c  = blockIdx.y;
    const int tb = blockIdx.x * TS;

    if (tid < TS) accS[tid] = 0.0f;

    for (int e = tid; e < ENTRY; e += BLOCK)
        mapl[e] = (unsigned short)tmap[c * ENTRY + e];

    const int gbase = tb * ENTRY;
    for (int i = tid; i < TS * ENTRY; i += BLOCK)
        sBits[i] = (unsigned char)samples[gbase + i];

    __syncthreads();

    for (int i = tid; i < TS * ENTRY; i += BLOCK) {
        const int m = i >> 10;
        const int e = i & (ENTRY - 1);
        mb[i] = sBits[(m << 10) | mapl[e]];
    }

    __syncthreads();

    v8i_t Bm;
#pragma unroll
    for (int v = 0; v < 8; ++v) {
        const int jb = 8 * (v >> 2) + 4 * half + (v & 3);
        Bm[v] = (jb == col) ? 0x01020408 : 0;
    }

    const int n0 = wid * 16;

    v8i_t A0, A1, A2, A3;
#pragma unroll
    for (int v = 0; v < 8; ++v) {
        const int j    = 8 * (v >> 2) + 4 * half + (v & 3);
        const int obase = (col << 10) + ((n0 + j) << 4);
        A0[v] = (int)*(const unsigned int*)(mb + obase + 0);
        A1[v] = (int)*(const unsigned int*)(mb + obase + 4);
        A2[v] = (int)*(const unsigned int*)(mb + obase + 8);
        A3[v] = (int)*(const unsigned int*)(mb + obase + 12);
    }

    const v8i_t zero = {0, 0, 0, 0, 0, 0, 0, 0};
    v8i_t accH = __builtin_amdgcn_wmma_i32_16x16x64_iu8(false, A0, false, Bm, zero,
                                                        false, false);
    v8i_t accL = __builtin_amdgcn_wmma_i32_16x16x64_iu8(false, A2, false, Bm, zero,
                                                        false, false);
#pragma unroll
    for (int r = 0; r < 8; ++r) accH[r] <<= 4;
#pragma unroll
    for (int r = 0; r < 8; ++r) accL[r] <<= 4;

    accH = __builtin_amdgcn_wmma_i32_16x16x64_iu8(false, A1, false, Bm, accH,
                                                  false, false);
    accL = __builtin_amdgcn_wmma_i32_16x16x64_iu8(false, A3, false, Bm, accL,
                                                  false, false);

    const int n = n0 + col;
    const float* ramRow = ram + (((size_t)c * NNEURONS + n) << ADDR_BITS);

    float vals[8];
#pragma unroll
    for (int r = 0; r < 8; ++r) {
        const unsigned int addr = (((unsigned)accH[r] << 8) + (unsigned)accL[r]) & 0xFFFF;
        vals[r] = ramRow[addr];
    }

#pragma unroll
    for (int r = 0; r < 8; ++r)
        atomicAdd(&accS[(half << 3) + r], vals[r]);

    __syncthreads();

    if (tid < TS)
        out[(size_t)(tb + tid) * NCLASSES + c] = accS[tid];
}

extern "C" void kernel_launch(void* const* d_in, const int* in_sizes, int n_in,
                              void* d_out, int out_size, void* d_ws, size_t ws_size,
                              hipStream_t stream)
{
    (void)n_in; (void)out_size; (void)d_ws; (void)ws_size;

    const int*   samples = (const int*)  d_in[0];
    const int*   tmap    = (const int*)  d_in[1];
    const float* ram     = (const float*)d_in[2];
    float*       out     = (float*)      d_out;

    const int nBatch = in_sizes[0] / ENTRY;
    dim3 grid(nBatch / TS, NCLASSES);
    wisard_wmma_kernel<<<grid, BLOCK, 0, stream>>>(samples, tmap, ram, out);
}
